// interpretable_multi_head_attention_66494683677220
// MI455X (gfx1250) — matrix operand order repaired and checked by reading, not compiled, not run on hardware
//
#include <hip/hip_runtime.h>
#include <hip/hip_bf16.h>

typedef __bf16 bf16;
typedef __attribute__((ext_vector_type(16))) __bf16 v16bf;
typedef __attribute__((ext_vector_type(8)))  float  v8f;
typedef __attribute__((ext_vector_type(4))) unsigned int u32x4;
typedef __attribute__((ext_vector_type(8))) int i32x8;
typedef __attribute__((ext_vector_type(4))) int i32x4;

static constexpr int Bc = 8;     // batch
static constexpr int Sc = 1024;  // sequence
static constexpr int Dc = 512;   // model dim
static constexpr int Hc = 8;     // heads
static constexpr int DKc = 64;   // head dim

#define WMMA_BF16(a, b, c)                                                     \
  __builtin_amdgcn_wmma_f32_16x16x32_bf16(false, (a), false, (b), (short)0,    \
                                          (c), false, false)

__device__ __forceinline__ unsigned rfl(unsigned x) {
  return (unsigned)__builtin_amdgcn_readfirstlane((int)x);
}

// ---------------------------------------------------------------------------
// TDM: 2D tile load Global -> LDS. Descriptor bitfields per CDNA5 ISA §8.3/8.4:
//   group0: [1:0]=count=1, [63:32]=lds_addr, [120:64]=global_addr, [127:126]=2
//   group1: [17:16]=data_size, [79:48]=tensor_dim0, [111:80]=tensor_dim1,
//           [127:112]=tile_dim0, [143:128]=tile_dim1, [207:160]=dim0_stride
// width/rows/stride in elements; ds_code: 1 = 2B, 2 = 4B.
// ---------------------------------------------------------------------------
__device__ __forceinline__ void tdm_load_2d(unsigned lds_addr, const void* gptr,
                                            unsigned width, unsigned rows,
                                            unsigned stride, unsigned ds_code) {
  unsigned long long ga = (unsigned long long)(uintptr_t)gptr;
  u32x4 g0 = {};
  g0[0] = rfl(1u);                                   // count=1, user mode
  g0[1] = rfl(lds_addr);                             // LDS byte address
  g0[2] = rfl((unsigned)(ga & 0xffffffffu));         // global addr lo
  g0[3] = rfl((unsigned)((ga >> 32) & 0x01ffffffu) | (2u << 30));  // hi | type=2
  i32x8 g1 = {};
  g1[0] = (int)rfl(ds_code << 16);                   // data_size
  g1[1] = (int)rfl(width << 16);                     // tensor_dim0 lo16 @48
  g1[2] = (int)rfl((width >> 16) | (rows << 16));    // dim0 hi | tensor_dim1 lo
  g1[3] = (int)rfl(((rows >> 16) & 0xffffu) | (width << 16));  // dim1 hi | tile_dim0
  g1[4] = (int)rfl(rows & 0xffffu);                  // tile_dim1 (tile_dim2=0)
  g1[5] = (int)rfl(stride);                          // tensor_dim0_stride lo32
  g1[6] = 0;
  g1[7] = 0;
  i32x4 z4 = {};
#if defined(__clang_major__) && (__clang_major__ >= 23)
  i32x8 z8 = {};
  __builtin_amdgcn_tensor_load_to_lds(g0, g1, z4, z4, z8, 0);
#else
  __builtin_amdgcn_tensor_load_to_lds(g0, g1, z4, z4, 0);
#endif
}

// ---------------------------------------------------------------------------
// Kernel 1: all input projections as bf16 GEMMs.
//   gemm id g: g<64 -> qh[h][b] = q[b] @ Wq[h]
//              g<128 -> kh[h][b] = k[b] @ Wk[h]
//              else  -> vh[b]    = v[b] @ Wv
// Each GEMM: [1024 x 512] x [512 x 64] -> bf16 [1024 x 64]
// Weight K-chunks (32x64 f32 = 8KB) are TDM-loaded to LDS per workgroup.
// ---------------------------------------------------------------------------
__global__ void __launch_bounds__(128)
proj_kernel(const float* __restrict__ q, const float* __restrict__ k,
            const float* __restrict__ v, const float* __restrict__ Wq,
            const float* __restrict__ Wk, const float* __restrict__ Wv,
            bf16* __restrict__ qh, bf16* __restrict__ kh,
            bf16* __restrict__ vh) {
  __shared__ float ldsW[32][64];

  const int g = blockIdx.y;
  const int mtile = blockIdx.x;  // 64 rows per block
  const int wave = threadIdx.x >> 5;
  const int lane = threadIdx.x & 31;
  const int lhalf = lane >> 4;
  const int l16 = lane & 15;

  const float* A;
  const float* W;
  bf16* C;
  if (g < 64) {
    int h = g >> 3, b = g & 7;
    A = q + (size_t)b * Sc * Dc;
    W = Wq + (size_t)h * Dc * DKc;
    C = qh + (size_t)(h * Bc + b) * Sc * DKc;
  } else if (g < 128) {
    int h = (g - 64) >> 3, b = (g - 64) & 7;
    A = k + (size_t)b * Sc * Dc;
    W = Wk + (size_t)h * Dc * DKc;
    C = kh + (size_t)(h * Bc + b) * Sc * DKc;
  } else {
    int b = g - 128;
    A = v + (size_t)b * Sc * Dc;
    W = Wv;
    C = vh + (size_t)b * Sc * DKc;
  }

  const int row = mtile * 64 + wave * 16 + l16;  // A row this lane feeds
  v8f acc[4] = {};

  for (int k0 = 0; k0 < Dc; k0 += 32) {
    __syncthreads();  // previous chunk fully consumed
    if (wave == 0) {
      tdm_load_2d((unsigned)(uintptr_t)&ldsW[0][0], W + (size_t)k0 * DKc,
                  /*width*/ 64, /*rows*/ 32, /*stride*/ 64, /*4B*/ 2);
      __builtin_amdgcn_s_wait_tensorcnt(0);
    }
    __syncthreads();  // chunk visible to all waves

    // A fragment (16x32 bf16): half 0 -> K 0-15, half 1 -> K 16-31
    v16bf a;
    {
      const float* ap = A + (size_t)row * Dc + k0 + 16 * lhalf;
#pragma unroll
      for (int e = 0; e < 16; ++e) a[e] = (bf16)ap[e];
    }
#pragma unroll
    for (int nt = 0; nt < 4; ++nt) {
      // B fragment (32x16) from LDS: lane = N, K = e + 16*half
      v16bf bfrag;
      const float* wp = &ldsW[16 * lhalf][nt * 16 + l16];
#pragma unroll
      for (int e = 0; e < 16; ++e) bfrag[e] = (bf16)wp[e * 64];
      acc[nt] = WMMA_BF16(a, bfrag, acc[nt]);
    }
  }

  // store C tile (C layout: row = r + 8*half, col = l16)
#pragma unroll
  for (int nt = 0; nt < 4; ++nt)
#pragma unroll
    for (int r = 0; r < 8; ++r) {
      int rr = mtile * 64 + wave * 16 + r + 8 * lhalf;
      C[(size_t)rr * DKc + nt * 16 + l16] = (bf16)acc[nt][r];
    }
}

// ---------------------------------------------------------------------------
// Kernel 2: fused attention. Workgroup = (b, 16-row query strip), loops heads.
//   scores -> exp (max-free) -> unnormalized write to attn -> row-sum reduce
//   -> P = e @ vh via WMMA (e staged per-wave through LDS; vh chunk TDM-loaded
//      into a per-wave LDS strip, overlapped with the score WMMAs)
//   -> in-place L2 rescale of the 64KB strip -> head-summed yacc in LDS.
// ---------------------------------------------------------------------------
__global__ void __launch_bounds__(128)
attn_kernel(const bf16* __restrict__ qh, const bf16* __restrict__ kh,
            const bf16* __restrict__ vh, const float* __restrict__ mask,
            float* __restrict__ attn_out, float* __restrict__ yacc) {
  __shared__ float ldsE[4][16][32];   // per-wave e-tile staging (C -> A relayout)
  __shared__ bf16 ldsV[4][32][64];    // per-wave vh chunk (TDM destination)
  __shared__ float ldsRowSum[16];
  __shared__ float ldsYacc[16][64];

  const int s0 = blockIdx.x * 16;
  const int b = blockIdx.y;
  const int wave = threadIdx.x >> 5;
  const int lane = threadIdx.x & 31;
  const int lhalf = lane >> 4;
  const int l16 = lane & 15;

  for (int i = threadIdx.x; i < 16 * 64; i += 128)
    (&ldsYacc[0][0])[i] = 0.0f;
  __syncthreads();

  const float* maskB = mask + (size_t)b * Sc * Sc;
  const bf16* vhb = vh + (size_t)b * Sc * DKc;

  for (int h = 0; h < Hc; ++h) {
    if (threadIdx.x < 16) ldsRowSum[threadIdx.x] = 0.0f;
    __syncthreads();

    const bf16* qhb = qh + ((size_t)(h * Bc + b) * Sc + s0) * DKc;
    const bf16* khb = kh + (size_t)(h * Bc + b) * Sc * DKc;
    float* arow = attn_out + ((size_t)(h * Bc + b) * Sc + s0) * Sc;

    // qh A-fragments for both K chunks (K=64 -> 2 x 32)
    v16bf aq[2];
#pragma unroll
    for (int kc = 0; kc < 2; ++kc) {
      const bf16* ap = qhb + (size_t)l16 * DKc + kc * 32 + 16 * lhalf;
#pragma unroll
      for (int e = 0; e < 16; ++e) aq[kc][e] = ap[e];
    }

    float sumpart[8] = {};
    v8f pacc[4] = {};
    const int t_begin = wave * 256;  // each wave owns a contiguous T range

    for (int tc = 0; tc < 8; ++tc) {  // 8 K-chunks of 32 keys
      // Per-wave TDM of this wave's vh chunk (TENSORcnt is per-wave -> no
      // cross-wave barrier). TDM is unordered vs DS: drain our LDS reads first.
      asm volatile("s_wait_dscnt 0x0" ::: "memory");
      tdm_load_2d((unsigned)(uintptr_t)&ldsV[wave][0][0],
                  vhb + (size_t)(t_begin + tc * 32) * DKc,
                  /*width*/ 64, /*rows*/ 32, /*stride*/ 64, /*2B*/ 1);
      // prefetch next kh strip while scores run
      __builtin_prefetch(khb + (size_t)(t_begin + tc * 32 + 32 + lane) * DKc,
                         0, 0);
#pragma unroll
      for (int j = 0; j < 2; ++j) {  // two 16x16 score tiles per chunk
        const int t0 = t_begin + tc * 32 + j * 16;
        v8f sc = {};
#pragma unroll
        for (int kc = 0; kc < 2; ++kc) {
          // kh^T B-fragment: column n = key t0+l16 (contiguous 16 bf16)
          const bf16* bp = khb + (size_t)(t0 + l16) * DKc + kc * 32 + 16 * lhalf;
          v16bf bk = *(const v16bf*)bp;
          sc = WMMA_BF16(aq[kc], bk, sc);
        }
#pragma unroll
        for (int r = 0; r < 8; ++r) {
          const int row = r + 8 * lhalf;
          const float m = maskB[(size_t)(s0 + row) * Sc + t0 + l16];
          const float e = __expf(sc[r] * 0.125f + (-1e9f) * (1.0f - m));
          sumpart[r] += e;
          arow[(size_t)row * Sc + t0 + l16] = e;  // unnormalized, via L2
          ldsE[wave][row][j * 16 + l16] = e;      // stage for P matmul
        }
      }
      // A-fragment of e (own wave's LDS strip; DS is in-order per wave)
      v16bf ae;
      {
        const float* ep = &ldsE[wave][l16][16 * lhalf];
#pragma unroll
        for (int e = 0; e < 16; ++e) ae[e] = (bf16)ep[e];
      }
      // vh chunk has landed by now (overlapped with score WMMAs)
      __builtin_amdgcn_s_wait_tensorcnt(0);
#pragma unroll
      for (int nt = 0; nt < 4; ++nt) {
        v16bf bv;
        const bf16* vp = &ldsV[wave][16 * lhalf][nt * 16 + l16];
#pragma unroll
        for (int e = 0; e < 16; ++e) bv[e] = vp[e * 64];
        pacc[nt] = WMMA_BF16(ae, bv, pacc[nt]);
      }
    }

    // reduce row sums across the 16 columns of each lane-half, then waves
#pragma unroll
    for (int r = 0; r < 8; ++r)
      for (int off = 1; off < 16; off <<= 1)
        sumpart[r] += __shfl_xor(sumpart[r], off, 16);
    if (l16 == 0) {
#pragma unroll
      for (int r = 0; r < 8; ++r)
        atomicAdd(&ldsRowSum[r + 8 * lhalf], sumpart[r]);
    }
    __syncthreads();  // row sums final; all e-stores visible in block

    // fold 1/rowsum into P and accumulate head sum into LDS
#pragma unroll
    for (int nt = 0; nt < 4; ++nt)
#pragma unroll
      for (int r = 0; r < 8; ++r) {
        const int row = r + 8 * lhalf;
        atomicAdd(&ldsYacc[row][nt * 16 + l16], pacc[nt][r] / ldsRowSum[row]);
      }

    // rescale the just-written 64KB strip in place (L2 turnaround)
    for (int base = threadIdx.x * 4; base < 16 * 1024; base += 128 * 4) {
      const int row = base >> 10;
      const int col = base & 1023;
      const float inv = 1.0f / ldsRowSum[row];
      float4* p = (float4*)(arow + (size_t)row * Sc + col);
      float4 x = *p;
      x.x *= inv; x.y *= inv; x.z *= inv; x.w *= inv;
      *p = x;
    }
    __syncthreads();  // protect ldsRowSum / ldsE reuse next head
  }

  // mean over heads -> yacc
  for (int i = threadIdx.x; i < 16 * 64; i += 128) {
    const int row = i >> 6, col = i & 63;
    yacc[((size_t)b * Sc + s0 + row) * DKc + col] =
        ldsYacc[row][col] * (1.0f / Hc);
  }
}

// ---------------------------------------------------------------------------
// Kernel 3: y = yacc @ Wo   ([8192 x 64] x [64 x 512])
// ---------------------------------------------------------------------------
__global__ void __launch_bounds__(128)
out_proj_kernel(const float* __restrict__ yacc, const float* __restrict__ Wo,
                float* __restrict__ y) {
  const int m0 = blockIdx.x * 16;
  const int n0 = blockIdx.y * 64 + (threadIdx.x >> 5) * 16;
  const int lane = threadIdx.x & 31;
  const int lhalf = lane >> 4;
  const int l16 = lane & 15;

  v8f acc = {};
#pragma unroll
  for (int kc = 0; kc < 2; ++kc) {
    v16bf a;
    {
      const float* ap = yacc + (size_t)(m0 + l16) * DKc + kc * 32 + 16 * lhalf;
#pragma unroll
      for (int e = 0; e < 16; ++e) a[e] = (bf16)ap[e];
    }
    v16bf bw;
    {
      const float* wp = Wo + (size_t)(kc * 32 + 16 * lhalf) * Dc + n0 + l16;
#pragma unroll
      for (int e = 0; e < 16; ++e) bw[e] = (bf16)wp[e * Dc];
    }
    acc = WMMA_BF16(a, bw, acc);
  }
#pragma unroll
  for (int r = 0; r < 8; ++r)
    y[(size_t)(m0 + r + 8 * lhalf) * Dc + n0 + l16] = acc[r];
}

// ---------------------------------------------------------------------------
extern "C" void kernel_launch(void* const* d_in, const int* in_sizes, int n_in,
                              void* d_out, int out_size, void* d_ws,
                              size_t ws_size, hipStream_t stream) {
  const float* q = (const float*)d_in[0];
  const float* k = (const float*)d_in[1];
  const float* v = (const float*)d_in[2];
  const float* mask = (const float*)d_in[3];
  const float* Wq = (const float*)d_in[4];
  const float* Wk = (const float*)d_in[5];
  const float* Wv = (const float*)d_in[6];
  const float* Wo = (const float*)d_in[7];

  float* y = (float*)d_out;                // [B,S,D]
  float* attn = y + (size_t)Bc * Sc * Dc;  // [H,B,S,S]

  char* ws = (char*)d_ws;
  bf16* qh = (bf16*)ws;                                   // 8 MiB
  bf16* kh = (bf16*)(ws + (size_t)8 * 1024 * 1024);       // 8 MiB
  bf16* vh = (bf16*)(ws + (size_t)16 * 1024 * 1024);      // 1 MiB
  float* yacc = (float*)(ws + (size_t)17 * 1024 * 1024);  // 2 MiB

  proj_kernel<<<dim3(Sc / 64, 136), 128, 0, stream>>>(q, k, v, Wq, Wk, Wv, qh,
                                                      kh, vh);
  attn_kernel<<<dim3(Sc / 16, Bc), 128, 0, stream>>>(qh, kh, vh, mask, attn,
                                                     yacc);
  out_proj_kernel<<<dim3((Bc * Sc) / 16, Dc / 64), 128, 0, stream>>>(yacc, Wo,
                                                                     y);
}
